// CIGAR_WO_PN_89026082111522
// MI455X (gfx1250) — compile-verified, matrix operand order checked by reading, not run on hardware
//
#include <hip/hip_runtime.h>

// ---------------- types & helpers ----------------
typedef __attribute__((ext_vector_type(16))) __bf16          v16bf;
typedef __attribute__((ext_vector_type(16))) unsigned short  v16u;
typedef __attribute__((ext_vector_type(8)))  unsigned short  v8u;
typedef __attribute__((ext_vector_type(8)))  float           v8f;

#define B_SZ 4096
#define S_SZ 200
#define N_NE 64
#define D_E  32
#define G_D  64

__device__ __forceinline__ unsigned short f2bf_u(float f) {
    union { float f; unsigned u; } v; v.f = f;
    unsigned r = v.u + 0x7FFFu + ((v.u >> 16) & 1u);   // RNE
    return (unsigned short)(r >> 16);
}

// branch-free tanh: v_exp_f32 is exp2 on AMD; 1 TRANS + ~6 VALU, no EXEC games.
// tanh(x) = sign(x) * (1 - t) / (1 + t),  t = 2^(-2*log2(e)*|x|)
__device__ __forceinline__ float fast_tanh(float x) {
    float ax = __builtin_fabsf(x);
    float t  = __builtin_amdgcn_exp2f(ax * -2.8853900817779268f);
    float r  = (1.0f - t) * __builtin_amdgcn_rcpf(1.0f + t);
    return __builtin_copysignf(r, x);
}

__device__ __forceinline__ v16bf frag_from_lds(const unsigned short* row, int base) {
    v8u lo = *(const v8u*)(row + base);
    v8u hi = *(const v8u*)(row + base + 16);
    v16u u = __builtin_shufflevector(lo, hi, 0,1,2,3,4,5,6,7,8,9,10,11,12,13,14,15);
    return __builtin_bit_cast(v16bf, u);
}
__device__ __forceinline__ v16bf frag_from_glb(const unsigned short* row, int base) {
    v8u lo = *(const v8u*)(row + base);
    v8u hi = *(const v8u*)(row + base + 16);
    v16u u = __builtin_shufflevector(lo, hi, 0,1,2,3,4,5,6,7,8,9,10,11,12,13,14,15);
    return __builtin_bit_cast(v16bf, u);
}

// ---------------- weight conversion ----------------
__global__ void cvt_f32_bf16(const float* __restrict__ src, unsigned short* __restrict__ dst, int n) {
    int i = blockIdx.x * blockDim.x + threadIdx.x;
    if (i < n) dst[i] = f2bf_u(src[i]);
}

// ---------------- GNN: per-sample 64x64 x 64x64 GEMM (x2 layers) ----------------
// block = 128 threads (4 waves), grid = B
__global__ void gnn_kernel(const int* __restrict__ neighbor_ids,
                           const float* __restrict__ mem0, const float* __restrict__ mem1,
                           const unsigned short* __restrict__ wagg0bf,
                           const unsigned short* __restrict__ wagg1bf,
                           const float* __restrict__ bias0, const float* __restrict__ bias1,
                           float* __restrict__ gnn_out) {
    __shared__ int            sNid[N_NE];
    __shared__ float          sMask[N_NE];
    __shared__ unsigned short sNeigh[N_NE][G_D];   // bf16 neighbor tile
    __shared__ unsigned short sW[G_D][G_D];        // bf16 W (row o, col d)
    __shared__ float          sSum[2][G_D];
    __shared__ float          sRDen;               // 1 / max(count,1)

    const int b    = blockIdx.x;
    const int t    = threadIdx.x;
    const int wave = t >> 5;
    const int lane = t & 31;

    if (t < N_NE) {
        int nid  = neighbor_ids[(size_t)b * N_NE + t];
        sNid[t]  = nid;
        sMask[t] = (nid != 0) ? 1.0f : 0.0f;
        sSum[0][t] = 0.0f;
        sSum[1][t] = 0.0f;
    }
    __syncthreads();
    if (t == 0) {
        float c = 0.0f;
        for (int i = 0; i < N_NE; ++i) c += sMask[i];
        sRDen = 1.0f / fmaxf(c, 1.0f);
    }

    for (int layer = 0; layer < 2; ++layer) {
        const float*          mem  = layer ? mem1 : mem0;
        const unsigned short* Wbf  = layer ? wagg1bf : wagg0bf;
        const float*          bias = layer ? bias1 : bias0;

        // stage W (64x64 bf16 = 2048 dwords) into LDS
        {
            const unsigned int* src = (const unsigned int*)Wbf;
            unsigned int*       dst = (unsigned int*)&sW[0][0];
            for (int i = 0; i < 16; ++i) dst[t + 128 * i] = src[t + 128 * i];
        }
        // gather neighbor rows, f32 -> bf16
        {
            const int l16 = t & 15;
            const int rw  = t >> 4;                 // 0..7
            for (int p = 0; p < 8; ++p) {
                int row = p * 8 + rw;
                int nid = sNid[row];
                float4 v = ((const float4*)(mem + (size_t)nid * G_D))[l16];
                sNeigh[row][l16 * 4 + 0] = f2bf_u(v.x);
                sNeigh[row][l16 * 4 + 1] = f2bf_u(v.y);
                sNeigh[row][l16 * 4 + 2] = f2bf_u(v.z);
                sNeigh[row][l16 * 4 + 3] = f2bf_u(v.w);
            }
        }
        __syncthreads();

        // wave 'wave' computes M-rows [16*wave, 16*wave+16), all 4 N-tiles
        v8f z = {0.f,0.f,0.f,0.f,0.f,0.f,0.f,0.f};
        v8f acc[4] = {z, z, z, z};
        const int m = wave * 16 + (lane & 15);
        for (int c = 0; c < 2; ++c) {
            const int base = 32 * c + ((lane >> 4) << 3);
            v16bf a = frag_from_lds(&sNeigh[m][0], base);
            #pragma unroll
            for (int nt = 0; nt < 4; ++nt) {
                const int o = nt * 16 + (lane & 15);
                v16bf bf = frag_from_lds(&sW[o][0], base);
                acc[nt] = __builtin_amdgcn_wmma_f32_16x16x32_bf16(
                    false, a, false, bf, (short)0, acc[nt], false, false);
            }
        }
        // epilogue: tanh, mask, reduce over neighbors
        #pragma unroll
        for (int nt = 0; nt < 4; ++nt) {
            const int o  = nt * 16 + (lane & 15);
            const float bo = bias[o];
            float partial = 0.0f;
            #pragma unroll
            for (int r = 0; r < 8; ++r) {
                int mm = wave * 16 + r + ((lane >> 4) << 3);
                partial += fast_tanh(acc[nt][r] + bo) * sMask[mm];
            }
            atomicAdd(&sSum[layer][o], partial);   // ds_add_f32
        }
        __syncthreads();
    }

    if (t < G_D) {
        float v = 0.5f * (sSum[0][t] + sSum[1][t]) * sRDen;
        gnn_out[(size_t)b * G_D + t] = v;
    }
}

// ---------------- feature build (seq masked mean fused) ----------------
// block = 64 threads, grid = B
__global__ void feat_kernel(const int* __restrict__ user_f0, const int* __restrict__ user_f1,
                            const int* __restrict__ adgroup_id, const int* __restrict__ cate_id,
                            const int* __restrict__ ad_seq, const int* __restrict__ cate_seq,
                            const float* __restrict__ user_tab0, const float* __restrict__ user_tab1,
                            const float* __restrict__ item_tab0, const float* __restrict__ item_tab1,
                            const float* __restrict__ gnn_out,
                            unsigned short* __restrict__ featbf) {
    __shared__ int sAd[S_SZ];
    __shared__ int sCa[S_SZ];
    const int b = blockIdx.x;
    const int t = threadIdx.x;              // 0..63

    for (int i = t; i < S_SZ; i += 64) {
        sAd[i] = ad_seq[(size_t)b * S_SZ + i];
        sCa[i] = cate_seq[(size_t)b * S_SZ + i];
    }
    __syncthreads();

    float sum = 0.0f, cnt = 0.0f;
    for (int s = 0; s < S_SZ; ++s) {
        int ad = sAd[s];
        float mk = (ad != 0) ? 1.0f : 0.0f;
        cnt += mk;
        float v = (t < 32) ? item_tab0[(size_t)ad * D_E + t]
                           : item_tab1[(size_t)sCa[s] * D_E + (t - 32)];
        sum += mk * v;
    }
    float mean = sum / fmaxf(cnt, 1.0f);

    float uemb = (t < 32) ? user_tab0[(size_t)user_f0[b] * D_E + t]
                          : user_tab1[(size_t)user_f1[b] * D_E + (t - 32)];
    float iemb = (t < 32) ? item_tab0[(size_t)adgroup_id[b] * D_E + t]
                          : item_tab1[(size_t)cate_id[b] * D_E + (t - 32)];
    float g = gnn_out[(size_t)b * G_D + t];

    unsigned short* f = featbf + (size_t)b * 320;
    f[      t] = f2bf_u(uemb);
    f[ 64 + t] = f2bf_u(iemb);
    f[128 + t] = f2bf_u(mean);
    f[192 + t] = f2bf_u(iemb * mean);
    f[256 + t] = f2bf_u(g);
}

// ---------------- generic bf16 WMMA GEMM: out = relu(A * W^T + bias) ----------------
// A: [M,K] bf16 row-major; W: [N,K] bf16 row-major. block = 128 (4 waves, 1 tile each)
__global__ void gemm_bias_relu(const unsigned short* __restrict__ A,
                               const unsigned short* __restrict__ W,
                               const float* __restrict__ bias,
                               unsigned short* __restrict__ outBf,
                               float* __restrict__ outF,
                               int M, int N, int K) {
    const int t    = threadIdx.x;
    const int wave = t >> 5;
    const int lane = t & 31;
    const int tilesN = N >> 4;
    const int tileId = blockIdx.x * 4 + wave;
    const int tm = tileId / tilesN;
    const int tn = tileId % tilesN;

    const unsigned short* arow = A + (size_t)(tm * 16 + (lane & 15)) * K;
    const unsigned short* wrow = W + (size_t)(tn * 16 + (lane & 15)) * K;

    v8f acc = {0.f,0.f,0.f,0.f,0.f,0.f,0.f,0.f};
    const int chunks = K >> 5;
    for (int c = 0; c < chunks; ++c) {
        const int base = 32 * c + ((lane >> 4) << 3);
        v16bf a  = frag_from_glb(arow, base);
        v16bf bf = frag_from_glb(wrow, base);
        acc = __builtin_amdgcn_wmma_f32_16x16x32_bf16(
            false, a, false, bf, (short)0, acc, false, false);
    }

    const int col = tn * 16 + (lane & 15);
    const float bo = bias[col];
    #pragma unroll
    for (int r = 0; r < 8; ++r) {
        int row = tm * 16 + r + ((lane >> 4) << 3);
        float v = fmaxf(acc[r] + bo, 0.0f);
        if (outBf) outBf[(size_t)row * N + col] = f2bf_u(v);
        if (outF)  outF [(size_t)row * N + col] = v;
    }
}

// ---------------- final 128-dot per row ----------------
// block = 128 (4 waves), one wave per output row
__global__ void final_kernel(const float* __restrict__ h2, const float* __restrict__ W3,
                             const float* __restrict__ b3, float* __restrict__ out) {
    const int t    = threadIdx.x;
    const int lane = t & 31;
    const int row  = blockIdx.x * 4 + (t >> 5);
    const float* h = h2 + (size_t)row * 128;
    float v = 0.0f;
    #pragma unroll
    for (int i = 0; i < 4; ++i) v += h[lane + 32 * i] * W3[lane + 32 * i];
    #pragma unroll
    for (int off = 16; off > 0; off >>= 1) v += __shfl_down(v, off, 32);
    if (lane == 0) out[row] = v + b3[0];
}

// ---------------- launch ----------------
extern "C" void kernel_launch(void* const* d_in, const int* in_sizes, int n_in,
                              void* d_out, int out_size, void* d_ws, size_t ws_size,
                              hipStream_t stream) {
    const int*   user_f0   = (const int*)d_in[0];
    const int*   user_f1   = (const int*)d_in[1];
    const int*   adgroup   = (const int*)d_in[2];
    const int*   cate      = (const int*)d_in[3];
    const int*   ad_seq    = (const int*)d_in[4];
    const int*   cate_seq  = (const int*)d_in[5];
    const int*   neigh     = (const int*)d_in[6];
    const float* user_tab0 = (const float*)d_in[7];
    const float* user_tab1 = (const float*)d_in[8];
    const float* item_tab0 = (const float*)d_in[9];
    const float* item_tab1 = (const float*)d_in[10];
    const float* mem0      = (const float*)d_in[11];
    const float* mem1      = (const float*)d_in[12];
    const float* W_agg0    = (const float*)d_in[13];
    const float* b_agg0    = (const float*)d_in[14];
    const float* W_agg1    = (const float*)d_in[15];
    const float* b_agg1    = (const float*)d_in[16];
    const float* W1        = (const float*)d_in[17];
    const float* b1        = (const float*)d_in[18];
    const float* W2        = (const float*)d_in[19];
    const float* b2        = (const float*)d_in[20];
    const float* W3        = (const float*)d_in[21];
    const float* b3        = (const float*)d_in[22];
    float* out = (float*)d_out;

    char* ws = (char*)d_ws;
    unsigned short* wagg0bf = (unsigned short*)(ws + 0);
    unsigned short* wagg1bf = (unsigned short*)(ws + 8192);
    unsigned short* w1bf    = (unsigned short*)(ws + 16384);
    unsigned short* w2bf    = (unsigned short*)(ws + 180224);
    float*          gnn_out = (float*)(ws + 245760);
    unsigned short* featbf  = (unsigned short*)(ws + 1294336);
    unsigned short* h1bf    = (unsigned short*)(ws + 3915776);
    float*          h2f     = (float*)(ws + 6012928);

    // 1) weight conversions
    cvt_f32_bf16<<<(4096 + 255) / 256, 256, 0, stream>>>(W_agg0, wagg0bf, 4096);
    cvt_f32_bf16<<<(4096 + 255) / 256, 256, 0, stream>>>(W_agg1, wagg1bf, 4096);
    cvt_f32_bf16<<<(81920 + 255) / 256, 256, 0, stream>>>(W1, w1bf, 81920);
    cvt_f32_bf16<<<(32768 + 255) / 256, 256, 0, stream>>>(W2, w2bf, 32768);

    // 2) GNN aggregation (WMMA)
    gnn_kernel<<<B_SZ, 128, 0, stream>>>(neigh, mem0, mem1, wagg0bf, wagg1bf,
                                         b_agg0, b_agg1, gnn_out);

    // 3) feature build (+ seq masked mean)
    feat_kernel<<<B_SZ, 64, 0, stream>>>(user_f0, user_f1, adgroup, cate,
                                         ad_seq, cate_seq,
                                         user_tab0, user_tab1, item_tab0, item_tab1,
                                         gnn_out, featbf);

    // 4) MLP GEMMs (WMMA): [4096,320]x[256,320]^T, then [4096,256]x[128,256]^T
    gemm_bias_relu<<<(B_SZ / 16) * (256 / 16) / 4, 128, 0, stream>>>(
        featbf, w1bf, b1, h1bf, nullptr, B_SZ, 256, 320);
    gemm_bias_relu<<<(B_SZ / 16) * (128 / 16) / 4, 128, 0, stream>>>(
        h1bf, w2bf, b2, nullptr, h2f, B_SZ, 128, 256);

    // 5) final dot + bias
    final_kernel<<<B_SZ / 4, 128, 0, stream>>>(h2f, W3, b3, out);
}
